// RPLayer_58042188038466
// MI455X (gfx1250) — 8-bit matrix operand order repaired and checked by reading, compiled, not run on hardware
//
#include <hip/hip_runtime.h>

typedef __attribute__((ext_vector_type(8))) int   v8i;
typedef __attribute__((ext_vector_type(8))) float v8f;

#define TB 64
#define HH 128
#define WW 128
#define CC 64
#define HW (HH * WW)
#define LDS_STRIDE 20

__device__ const unsigned char kF8E4M3[16] = {
    0x00, 0x38, 0x40, 0x44, 0x48, 0x4A, 0x4C, 0x4E,
    0x50, 0x51, 0x52, 0x53, 0x54, 0x55, 0x56, 0x57};

__global__ void rp_prep_kernel(const int* __restrict__ rp_map,
                               unsigned* __restrict__ wsA) {
    const int lane = threadIdx.x & 31;
    const int ct   = threadIdx.x >> 5;
    const int m    = lane & 15;
    const int c    = ct * 16 + m;
    const int hi   = lane >> 4;
    const int r0 = rp_map[c * 4 + 0];
    const int r1 = rp_map[c * 4 + 1];
    const int r2 = rp_map[c * 4 + 2];
    const int r3 = rp_map[c * 4 + 3];
#pragma unroll
    for (int v = 0; v < 8; ++v) {
        const int kbase = (v >> 2) * 32 + hi * 16 + (v & 3) * 4;
        unsigned word = 0u;
#pragma unroll
        for (int byt = 0; byt < 4; ++byt) {
            const int k = kbase + byt;
            unsigned val = 0u;
            if (r0 == k) val += 1u;
            if (r1 == k) val += 2u;
            if (r2 == k) val += 4u;
            if (r3 == k) val += 8u;
            word |= (unsigned)kF8E4M3[val] << (8 * byt);
        }
        wsA[(ct * 32 + lane) * 8 + v] = word;
    }
}

__global__ void __launch_bounds__(256)
rp_main_kernel(const float* __restrict__ x,
               const unsigned* __restrict__ wsA,
               float* __restrict__ out) {
    __shared__ unsigned lds[WW * LDS_STRIDE];

    const int g    = blockIdx.x;
    const int b    = g >> 7;
    const int h    = g & 127;
    const int tid  = threadIdx.x;
    const int lane = tid & 31;

    const float* xb = x + (size_t)b * TB * HW + (size_t)h * WW;

    v8i A[4];
#pragma unroll
    for (int ct = 0; ct < 4; ++ct)
        A[ct] = *reinterpret_cast<const v8i*>(wsA + (ct * 32 + lane) * 8);

#pragma unroll
    for (int it = 0; it < 2; ++it) {
        const int idx = tid + it * 256;
        const int wq  = idx & 31;
        const int kq  = idx >> 5;
        const uint4 t0 = *reinterpret_cast<const uint4*>(xb + (size_t)(4 * kq + 0) * HW + 4 * wq);
        const uint4 t1 = *reinterpret_cast<const uint4*>(xb + (size_t)(4 * kq + 1) * HW + 4 * wq);
        const uint4 t2 = *reinterpret_cast<const uint4*>(xb + (size_t)(4 * kq + 2) * HW + 4 * wq);
        const uint4 t3 = *reinterpret_cast<const uint4*>(xb + (size_t)(4 * kq + 3) * HW + 4 * wq);
        const unsigned w0 = ((t0.x >> 23) & 0x38u)       | ((t1.x >> 15) & 0x3800u)
                          | ((t2.x >> 7)  & 0x380000u)   | ((t3.x << 1)  & 0x38000000u);
        const unsigned w1 = ((t0.y >> 23) & 0x38u)       | ((t1.y >> 15) & 0x3800u)
                          | ((t2.y >> 7)  & 0x380000u)   | ((t3.y << 1)  & 0x38000000u);
        const unsigned w2 = ((t0.z >> 23) & 0x38u)       | ((t1.z >> 15) & 0x3800u)
                          | ((t2.z >> 7)  & 0x380000u)   | ((t3.z << 1)  & 0x38000000u);
        const unsigned w3 = ((t0.w >> 23) & 0x38u)       | ((t1.w >> 15) & 0x3800u)
                          | ((t2.w >> 7)  & 0x380000u)   | ((t3.w << 1)  & 0x38000000u);
        lds[(4 * wq + 0) * LDS_STRIDE + kq] = w0;
        lds[(4 * wq + 1) * LDS_STRIDE + kq] = w1;
        lds[(4 * wq + 2) * LDS_STRIDE + kq] = w2;
        lds[(4 * wq + 3) * LDS_STRIDE + kq] = w3;
    }
    __syncthreads();

    const int wt  = tid >> 5;
    const int n   = lane & 15;
    const int hi  = lane >> 4;
    const int row = wt * 16 + n;

    const uint4 blo = *reinterpret_cast<const uint4*>(&lds[row * LDS_STRIDE + hi * 4]);
    const uint4 bhi = *reinterpret_cast<const uint4*>(&lds[row * LDS_STRIDE + 8 + hi * 4]);
    v8i Bf;
    Bf[0] = (int)blo.x; Bf[1] = (int)blo.y; Bf[2] = (int)blo.z; Bf[3] = (int)blo.w;
    Bf[4] = (int)bhi.x; Bf[5] = (int)bhi.y; Bf[6] = (int)bhi.z; Bf[7] = (int)bhi.w;

    v8f acc[4];
#pragma unroll
    for (int ct = 0; ct < 4; ++ct) {
        v8f z = {0.f, 0.f, 0.f, 0.f, 0.f, 0.f, 0.f, 0.f};
        acc[ct] = __builtin_amdgcn_wmma_f32_16x16x64_fp8_fp8(
            A[ct], Bf,  (short)0, z,
             false,  false);
    }

    const int w = wt * 16 + n;
    float* obw = out + (size_t)b * CC * HW + (size_t)(hi * 8) * HW + (size_t)h * WW + w;
#pragma unroll
    for (int ct = 0; ct < 4; ++ct)
#pragma unroll
        for (int r = 0; r < 8; ++r)
            obw[(size_t)(ct * 16 + r) * HW] = acc[ct][r];
}

extern "C" void kernel_launch(void* const* d_in, const int* in_sizes, int n_in,
                              void* d_out, int out_size, void* d_ws, size_t ws_size,
                              hipStream_t stream) {
    (void)in_sizes; (void)n_in; (void)out_size; (void)ws_size;
    const float* x   = (const float*)d_in[0];
    const int*   rp  = (const int*)d_in[1];
    unsigned*    wsA = (unsigned*)d_ws;
    float*       out = (float*)d_out;

    rp_prep_kernel<<<1, 128, 0, stream>>>(rp, wsA);
    rp_main_kernel<<<16 * 128, 256, 0, stream>>>(x, wsA, out);
}
